// DFlashAttentionV5_72292889526399
// MI455X (gfx1250) — matrix operand order repaired and checked by reading, not compiled, not run on hardware
//
#include <hip/hip_runtime.h>
#include <hip/hip_bf16.h>

typedef __attribute__((ext_vector_type(16))) __bf16 v16bf;
typedef __attribute__((ext_vector_type(8)))  float  v8f;
typedef __attribute__((ext_vector_type(4)))  unsigned int u32x4;
typedef __attribute__((ext_vector_type(8)))  int          i32x8;
typedef __attribute__((ext_vector_type(4)))  int          i32x4;
typedef unsigned short u16;

// ---------------- helpers ----------------

__device__ inline u16 f2bf(float f) {
    union { float f; unsigned u; } v; v.f = f;
    unsigned r = v.u + 0x7FFFu + ((v.u >> 16) & 1u);   // round-to-nearest-even
    return (u16)(r >> 16);
}

// A-fragment (16x32 bf16, row-major source, ld in elements).
// Per ISA 7.12.2: lane L (m = L&15, hi = L>>4) holds K = hi*8+0..7 in halves 0..7
// and K = 16+hi*8+0..7 in halves 8..15  -> two contiguous 16B chunks.
__device__ inline v16bf load_frag_a(const u16* p, long ld, int lane) {
    const int row = lane & 15, hi = (lane >> 4) & 1;
    const u16* s = p + (long)row * ld + hi * 8;
    union { v16bf v; uint4 q[2]; } f;
    f.q[0] = *(const uint4*)(s);
    f.q[1] = *(const uint4*)(s + 16);
    return f.v;
}

// B-fragment (32x16 bf16). Source stored "column-major" (each output column's K run
// is contiguous, stride ld between columns). lane L (n = L&15, hi = L>>4) holds
// K = hi*8+0..7 in halves 0..7 and K = 16+hi*8+0..7 in halves 8..15 -> two 16B chunks.
__device__ inline v16bf load_frag_b(const u16* p, long ld, int lane) {
    const int col = lane & 15, hi = (lane >> 4) & 1;
    const u16* s = p + (long)col * ld + hi * 8;
    union { v16bf v; uint4 q[2]; } f;
    f.q[0] = *(const uint4*)(s);
    f.q[1] = *(const uint4*)(s + 16);
    return f.v;
}

__device__ inline v8f wmma_bf16(v16bf a, v16bf b, v8f c) {
    return __builtin_amdgcn_wmma_f32_16x16x32_bf16(false, a, false, b, (short)0, c, false, false);
}

__device__ inline v8f zero_v8f() {
    v8f z = {0.f, 0.f, 0.f, 0.f, 0.f, 0.f, 0.f, 0.f};
    return z;
}

// ---------------- Tensor Data Mover: 2-D tile (tile_x elems wide, 128 rows) ----------------
// Builds the D# per ISA ch.8 and posts TENSOR_LOAD_TO_LDS. Rows land packed in LDS
// (row y at lds_off + y*tile_x*2). Tracked by TENSORcnt.
__device__ inline void tdm_load_tile_128(const u16* gsrc, unsigned lds_off,
                                         unsigned tile_x, unsigned tensor_d0,
                                         long row_stride_elems) {
    unsigned long long ga = (unsigned long long)(size_t)gsrc;
    u32x4 g0;
    g0[0] = 1u;                                             // count=1, user mode
    g0[1] = lds_off;                                        // LDS byte address
    g0[2] = (unsigned)(ga & 0xFFFFFFFFu);                   // global_addr[31:0]
    g0[3] = (unsigned)((ga >> 32) & 0x01FFFFFFu) | (2u << 30); // addr[56:32] | type=2
    i32x8 g1;
    g1[0] = (int)(1u << 16);                                // data_size=1 (2 bytes)
    g1[1] = (int)((tensor_d0 & 0xFFFFu) << 16);             // tensor_dim0[15:0] @bits63:48
    g1[2] = (int)(((tensor_d0 >> 16) & 0xFFFFu) | (128u << 16)); // dim0 hi | tensor_dim1=128
    g1[3] = (int)(tile_x << 16);                            // tensor_dim1 hi=0 | tile_dim0
    g1[4] = 128;                                            // tile_dim1=128, tile_dim2=0
    unsigned long long st = (unsigned long long)row_stride_elems;
    g1[5] = (int)(unsigned)(st & 0xFFFFFFFFu);              // tensor_dim0_stride[31:0]
    g1[6] = (int)(unsigned)((st >> 32) & 0xFFFFu);          // stride hi | dim1_stride lo=0
    g1[7] = 0;
    i32x4 z4 = {0, 0, 0, 0};
#if defined(__clang_major__) && (__clang_major__ >= 23)
    i32x8 z8 = {0, 0, 0, 0, 0, 0, 0, 0};
    __builtin_amdgcn_tensor_load_to_lds(g0, g1, z4, z4, z8, 0);
#else
    __builtin_amdgcn_tensor_load_to_lds(g0, g1, z4, z4, 0);
#endif
}

// ---------------- elementwise converts ----------------

__global__ __launch_bounds__(256)
void convert_bf16_kernel(const float* __restrict__ in, u16* __restrict__ out, long n4) {
    long i = (long)blockIdx.x * 256 + threadIdx.x;
    if (i >= n4) return;
    float4 v = ((const float4*)in)[i];
    union { u16 h[4]; uint2 q; } r;
    r.h[0] = f2bf(v.x); r.h[1] = f2bf(v.y); r.h[2] = f2bf(v.z); r.h[3] = f2bf(v.w);
    ((uint2*)out)[i] = r.q;
}

// out[c*R + r] = bf16(in[r*C + c]) ; 32x32 LDS tile, block = 256 (32x8)
__global__ __launch_bounds__(256)
void transpose_bf16_kernel(const float* __restrict__ in, u16* __restrict__ out, int R, int C) {
    __shared__ float tile[32][33];
    int c0 = blockIdx.x * 32, r0 = blockIdx.y * 32;
    int tx = threadIdx.x & 31, ty = threadIdx.x >> 5;
#pragma unroll
    for (int k = 0; k < 32; k += 8)
        tile[ty + k][tx] = in[(long)(r0 + ty + k) * C + c0 + tx];
    __syncthreads();
#pragma unroll
    for (int k = 0; k < 32; k += 8)
        out[(long)(c0 + ty + k) * R + r0 + tx] = f2bf(tile[tx][ty + k]);
}

// ---------------- bf16 WMMA GEMM, TDM-staged: C[M,N] f32 = A[M,K] * Bt[N,K]^T --------------
// block = 256 threads = 8 waves (2 in M x 4 in N), block tile 128x128.
// Per 64-wide K step the TDM DMAs A(128x64) and B(128x64) bf16 tiles into
// double-buffered LDS (64 KB); wave 0 posts the descriptors and gates the block
// with s_wait_tensorcnt; all waves then feed WMMA from ds_load_b128.

__global__ __launch_bounds__(256)
void gemm_bf16_nt_kernel(const u16* __restrict__ A, const u16* __restrict__ Bt,
                         float* __restrict__ C, int M, int N, int Kd) {
    __shared__ __align__(16) u16 ldsA[2][128 * 64];
    __shared__ __align__(16) u16 ldsB[2][128 * 64];

    int lane = threadIdx.x & 31;
    int wave = threadIdx.x >> 5;
    int wm = wave & 1, wn = wave >> 1;
    int bm = blockIdx.y * 128;
    int bn = blockIdx.x * 128;
    int col = lane & 15, hi = lane >> 4;

    v8f acc[4][2];
#pragma unroll
    for (int i = 0; i < 4; ++i)
#pragma unroll
        for (int j = 0; j < 2; ++j) acc[i][j] = zero_v8f();

    const u16* Ab = A  + (long)bm * Kd;
    const u16* Bb = Bt + (long)bn * Kd;
    const int KT = Kd >> 6;   // 64-wide K steps

    if (wave == 0) {   // prologue: fill buffer 0
        tdm_load_tile_128(Ab, (unsigned)(size_t)&ldsA[0][0], 64, (unsigned)Kd, Kd);
        tdm_load_tile_128(Bb, (unsigned)(size_t)&ldsB[0][0], 64, (unsigned)Kd, Kd);
    }

    for (int kt = 0; kt < KT; ++kt) {
        int cur = kt & 1, nxt = cur ^ 1;
        __syncthreads();                        // buf[nxt] no longer being read
        if (wave == 0) {
            if (kt + 1 < KT) {
                long k0 = (long)(kt + 1) << 6;
                tdm_load_tile_128(Ab + k0, (unsigned)(size_t)&ldsA[nxt][0], 64, (unsigned)Kd, Kd);
                tdm_load_tile_128(Bb + k0, (unsigned)(size_t)&ldsB[nxt][0], 64, (unsigned)Kd, Kd);
                __builtin_amdgcn_s_wait_tensorcnt(2);   // buf[cur] pair has landed
            } else {
                __builtin_amdgcn_s_wait_tensorcnt(0);
            }
        }
        __syncthreads();                        // release compute on buf[cur]
        const u16* As = &ldsA[cur][0];
        const u16* Bs = &ldsB[cur][0];
#pragma unroll
        for (int d = 0; d < 2; ++d) {
            v16bf a[4], b[2];
#pragma unroll
            for (int i = 0; i < 4; ++i)
                a[i] = load_frag_a(As + (long)(wm * 64 + 16 * i) * 64 + 32 * d, 64, lane);
#pragma unroll
            for (int j = 0; j < 2; ++j)
                b[j] = load_frag_b(Bs + (long)(wn * 32 + 16 * j) * 64 + 32 * d, 64, lane);
#pragma unroll
            for (int i = 0; i < 4; ++i)
#pragma unroll
                for (int j = 0; j < 2; ++j)
                    acc[i][j] = wmma_bf16(a[i], b[j], acc[i][j]);
        }
    }
    // C layout: vgpr r, lanes 0-15 -> M=r, lanes 16-31 -> M=r+8, N = lane&15
#pragma unroll
    for (int i = 0; i < 4; ++i)
#pragma unroll
        for (int j = 0; j < 2; ++j)
#pragma unroll
            for (int r = 0; r < 8; ++r)
                C[(long)(bm + wm * 64 + 16 * i + r + 8 * hi) * N
                  + bn + wn * 32 + 16 * j + col] = acc[i][j][r];
}

// ---------------- RMS-norm + gather into [B,H,Lout,HD] bf16 ----------------

__global__ __launch_bounds__(256)
void rmsnorm_gather_kernel(const float* __restrict__ Y, const float* __restrict__ w,
                           u16* __restrict__ out, int S, int Lout, int seq_off) {
    const int H = 16, HD = 128;
    int lane = threadIdx.x & 31;
    long wg = (long)blockIdx.x * 8 + (threadIdx.x >> 5);
    if (wg >= (long)2 * S * H) return;
    int h = (int)(wg % H);
    long bs = wg / H;                 // b*S + s
    int s = (int)(bs % S);
    int b = (int)(bs / S);

    const float* src = Y + bs * (H * HD) + h * HD;
    float4 v = ((const float4*)src)[lane];
    float ss = v.x * v.x + v.y * v.y + v.z * v.z + v.w * v.w;
#pragma unroll
    for (int m = 1; m < 32; m <<= 1) ss += __shfl_xor(ss, m, 32);
    float inv = rsqrtf(ss * (1.0f / 128.0f) + 1e-6f);
    float4 wv = ((const float4*)w)[lane];

    u16* dst = out + ((long)(b * H + h) * Lout + seq_off + s) * HD + lane * 4;
    union { u16 h4[4]; uint2 q; } r;
    r.h4[0] = f2bf(v.x * inv * wv.x);
    r.h4[1] = f2bf(v.y * inv * wv.y);
    r.h4[2] = f2bf(v.z * inv * wv.z);
    r.h4[3] = f2bf(v.w * inv * wv.w);
    *(uint2*)dst = r.q;
}

// V gather + transpose: out[b][h][hd][seq_off+s] = bf16(Y[i]) with Y [(B*S), H*HD]
__global__ __launch_bounds__(256)
void vt_gather_kernel(const float* __restrict__ Y, u16* __restrict__ out,
                      int S, int Lfull, int seq_off) {
    const int H = 16, HD = 128;
    long i = (long)blockIdx.x * 256 + threadIdx.x;
    if (i >= (long)2 * S * H * HD) return;
    int hd = (int)(i % HD);
    long t = i / HD;
    int h = (int)(t % H); t /= H;     // t = b*S + s
    int s = (int)(t % S);
    int b = (int)(t / S);
    out[((long)(b * H + h) * HD + hd) * Lfull + seq_off + s] = f2bf(Y[i]);
}

// ---------------- flash attention ----------------
// grid (B*H, Kq/128); block 256 = 8 waves; each wave owns 16 q rows.
// Per 32-wide KV step: 8 WMMAs for S = Q*K^T, softmax update in C-layout,
// P re-laid out via LDS into A-fragment layout, 8 WMMAs for O += P*V.

__global__ __launch_bounds__(256)
void flash_attn_kernel(const u16* __restrict__ Qn, const u16* __restrict__ Kn,
                       const u16* __restrict__ Vt, const float* __restrict__ mask,
                       u16* __restrict__ attn_out) {
    const int H = 16, HD = 128, Kq = 1024, L = 3072;
    __shared__ __align__(16) u16 plds[8][16 * 32];

    int lane = threadIdx.x & 31;
    int wave = threadIdx.x >> 5;
    int col = lane & 15, hi = lane >> 4;
    int b = blockIdx.x >> 4, h = blockIdx.x & 15;
    int q0 = blockIdx.y * 128 + wave * 16;

    const u16* Qh = Qn + ((long)(b * H + h) * Kq + q0) * HD;   // 16 x 128 row-major
    const u16* Kh = Kn + (long)(b * H + h) * L * HD;           // L x 128 row-major
    const u16* Vh = Vt + (long)(b * H + h) * HD * L;           // 128 x L (hd-major)

    v16bf qf[4];
#pragma unroll
    for (int d = 0; d < 4; ++d) qf[d] = load_frag_a(Qh + d * 32, HD, lane);

    v8f o[8];
#pragma unroll
    for (int j = 0; j < 8; ++j) o[j] = zero_v8f();
    float m_i[8], l_i[8];
#pragma unroll
    for (int r = 0; r < 8; ++r) { m_i[r] = -3.0e38f; l_i[r] = 0.f; }
    const float scale = 0.08838834764831845f;   // 1/sqrt(128)

    for (int l0 = 0; l0 < L; l0 += 32) {
        v8f s0 = zero_v8f(), s1 = zero_v8f();
#pragma unroll
        for (int d = 0; d < 4; ++d) {
            v16bf kb0 = load_frag_b(Kh + (long)l0 * HD + d * 32, HD, lane);
            v16bf kb1 = load_frag_b(Kh + (long)(l0 + 16) * HD + d * 32, HD, lane);
            s0 = wmma_bf16(qf[d], kb0, s0);
            s1 = wmma_bf16(qf[d], kb1, s1);
        }
#pragma unroll
        for (int r = 0; r < 8; ++r) {
            int qrow = q0 + r + 8 * hi;
            const float* mp = mask + ((long)b * Kq + qrow) * L + l0 + col;
            float sv0 = s0[r] * scale + mp[0];
            float sv1 = s1[r] * scale + mp[16];
            float mx = fmaxf(sv0, sv1);
#pragma unroll
            for (int m = 1; m < 16; m <<= 1) mx = fmaxf(mx, __shfl_xor(mx, m, 32));
            float mnew = fmaxf(m_i[r], mx);
            float alpha = __expf(m_i[r] - mnew);
            float p0 = __expf(sv0 - mnew);
            float p1 = __expf(sv1 - mnew);
            float rs = p0 + p1;
#pragma unroll
            for (int m = 1; m < 16; m <<= 1) rs += __shfl_xor(rs, m, 32);
            l_i[r] = l_i[r] * alpha + rs;
            m_i[r] = mnew;
#pragma unroll
            for (int j = 0; j < 8; ++j) o[j][r] *= alpha;
            plds[wave][(r + 8 * hi) * 32 + col]      = f2bf(p0);
            plds[wave][(r + 8 * hi) * 32 + 16 + col] = f2bf(p1);
        }
        asm volatile("s_wait_dscnt 0x0" ::: "memory");   // intra-wave LDS RAW
        v16bf pa = load_frag_a(&plds[wave][0], 32, lane);
#pragma unroll
        for (int j = 0; j < 8; ++j) {
            v16bf vb = load_frag_b(Vh + (long)(16 * j) * L + l0, L, lane);
            o[j] = wmma_bf16(pa, vb, o[j]);
        }
    }
    // epilogue: normalize, write attn [B*Kq, H*HD] bf16
#pragma unroll
    for (int r = 0; r < 8; ++r) {
        float inv = 1.0f / l_i[r];
        long row = (long)b * Kq + q0 + r + 8 * hi;
#pragma unroll
        for (int j = 0; j < 8; ++j)
            attn_out[row * (H * HD) + h * HD + 16 * j + col] = f2bf(o[j][r] * inv);
    }
}

// ---------------- host launcher ----------------

extern "C" void kernel_launch(void* const* d_in, const int* in_sizes, int n_in,
                              void* d_out, int out_size, void* d_ws, size_t ws_size,
                              hipStream_t stream) {
    const int B = 2, Kq = 1024, CTX = 2048, D = 2048, H = 16, HD = 128;
    const int L = CTX + Kq;                 // 3072
    (void)in_sizes; (void)n_in; (void)out_size; (void)ws_size;

    const float* x      = (const float*)d_in[0];
    const float* ctx    = (const float*)d_in[1];
    const float* mask   = (const float*)d_in[2];
    const float* w_src[6] = { (const float*)d_in[3], (const float*)d_in[4],
                              (const float*)d_in[5], (const float*)d_in[6],
                              (const float*)d_in[7], (const float*)d_in[8] };
    const float* q_norm_w = (const float*)d_in[9];
    const float* k_norm_w = (const float*)d_in[10];

    char* ws = (char*)d_ws;
    size_t off = 0;
    auto alloc = [&](size_t bytes) -> void* {
        void* p = ws + off;
        off += (bytes + 255) & ~(size_t)255;
        return p;
    };

    u16* xb = (u16*)alloc((size_t)B * Kq * D * 2);
    u16* cb = (u16*)alloc((size_t)B * CTX * D * 2);
    u16* wT[6];
    for (int i = 0; i < 6; ++i) wT[i] = (u16*)alloc((size_t)D * D * 2);
    float* Yq  = (float*)alloc((size_t)B * Kq * D * 4);
    float* Yk  = (float*)alloc((size_t)B * Kq * D * 4);
    float* Yv  = (float*)alloc((size_t)B * Kq * D * 4);
    float* Yck = (float*)alloc((size_t)B * CTX * D * 4);
    float* Ycv = (float*)alloc((size_t)B * CTX * D * 4);
    u16* Qn   = (u16*)alloc((size_t)B * H * Kq * HD * 2);
    u16* Kn   = (u16*)alloc((size_t)B * H * L * HD * 2);
    u16* Vt   = (u16*)alloc((size_t)B * H * L * HD * 2);
    u16* attn = (u16*)alloc((size_t)B * Kq * D * 2);

    // 1) fp32 -> bf16 activations
    long n4x = (long)B * Kq * D / 4;
    convert_bf16_kernel<<<dim3((unsigned)((n4x + 255) / 256)), 256, 0, stream>>>(x, xb, n4x);
    long n4c = (long)B * CTX * D / 4;
    convert_bf16_kernel<<<dim3((unsigned)((n4c + 255) / 256)), 256, 0, stream>>>(ctx, cb, n4c);

    // 2) transpose+convert weights to bf16 [N,K]
    dim3 tg(D / 32, D / 32);
    for (int i = 0; i < 6; ++i)
        transpose_bf16_kernel<<<tg, 256, 0, stream>>>(w_src[i], wT[i], D, D);

    // 3) projection GEMMs (TDM-staged)
    dim3 gq(D / 128, (B * Kq) / 128);    // (16,16)
    dim3 gc(D / 128, (B * CTX) / 128);   // (16,32)
    gemm_bf16_nt_kernel<<<gq, 256, 0, stream>>>(xb, wT[0], Yq,  B * Kq,  D, D);
    gemm_bf16_nt_kernel<<<gq, 256, 0, stream>>>(xb, wT[1], Yk,  B * Kq,  D, D);
    gemm_bf16_nt_kernel<<<gq, 256, 0, stream>>>(xb, wT[2], Yv,  B * Kq,  D, D);
    gemm_bf16_nt_kernel<<<gc, 256, 0, stream>>>(cb, wT[3], Yck, B * CTX, D, D);
    gemm_bf16_nt_kernel<<<gc, 256, 0, stream>>>(cb, wT[4], Ycv, B * CTX, D, D);

    // 4) RMS-norm + gather heads (Q; K = concat[ctx, self]); V transpose-gather
    rmsnorm_gather_kernel<<<(2 * Kq  * H) / 8, 256, 0, stream>>>(Yq,  q_norm_w, Qn, Kq,  Kq, 0);
    rmsnorm_gather_kernel<<<(2 * CTX * H) / 8, 256, 0, stream>>>(Yck, k_norm_w, Kn, CTX, L,  0);
    rmsnorm_gather_kernel<<<(2 * Kq  * H) / 8, 256, 0, stream>>>(Yk,  k_norm_w, Kn, Kq,  L,  CTX);
    vt_gather_kernel<<<(unsigned)(((long)2 * CTX * H * HD) / 256), 256, 0, stream>>>(Ycv, Vt, CTX, L, 0);
    vt_gather_kernel<<<(unsigned)(((long)2 * Kq  * H * HD) / 256), 256, 0, stream>>>(Yv,  Vt, Kq,  L, CTX);

    // 5) flash attention
    dim3 ag(B * H, Kq / 128);
    flash_attn_kernel<<<ag, 256, 0, stream>>>(Qn, Kn, Vt, mask, attn);

    // 6) output projection -> d_out (fp32)
    gemm_bf16_nt_kernel<<<gq, 256, 0, stream>>>(attn, wT[5], (float*)d_out, B * Kq, D, D);
}
